// WindowAttention_1382979470056
// MI455X (gfx1250) — matrix operand order repaired and checked by reading, not compiled, not run on hardware
//
#include <hip/hip_runtime.h>
#include <hip/hip_bf16.h>

typedef __attribute__((ext_vector_type(16))) __bf16 v16bf;
typedef __attribute__((ext_vector_type(8)))  __bf16 v8bf;
typedef __attribute__((ext_vector_type(4)))  __bf16 v4bf;
typedef __attribute__((ext_vector_type(8)))  float  v8f;
typedef __attribute__((ext_vector_type(4)))  float  v4f;

#define N_HEADS 6
#define NTOK    49
#define EDIM    192
#define HDIM    32
#define NPAD    64
#define SPITCH  68   // padded f32 pitch for score buffer: bank = (4m+j)%64, conflict-free

// ---------------- fragment loaders (bf16, 16x16x32 WMMA) ----------------
// A: 16x32 (MxK). lane L: m = m0 + L%16 ; K runs {g*8..+7} and {16+g*8..+7}, g=L/16.
__device__ __forceinline__ v16bf load_A_frag(const __bf16* base, int m0, int k0, int pitch, int lane) {
    const int g   = lane >> 4;
    const int row = m0 + (lane & 15);
    const __bf16* p = base + row * pitch + k0 + g * 8;
    v8bf lo = *(const v8bf*)(p);
    v8bf hi = *(const v8bf*)(p + 16);
    v16bf a;
#pragma unroll
    for (int i = 0; i < 8; ++i) { a[i] = lo[i]; a[8 + i] = hi[i]; }
    return a;
}

// B: 32x16 (KxN), stored N-major (base[n*pitch + k]). lane L: n = n0 + L%16 ;
// K runs {g*8..+7} and {16+g*8..+7} (relative to k0), g = L/16.
__device__ __forceinline__ v16bf load_B_frag(const __bf16* base, int n0, int k0, int pitch, int lane) {
    const int g = lane >> 4;
    const int n = n0 + (lane & 15);
    const __bf16* p = base + n * pitch + k0 + g * 8;
    v8bf lo = *(const v8bf*)(p);
    v8bf hi = *(const v8bf*)(p + 16);
    v16bf b;
#pragma unroll
    for (int i = 0; i < 8; ++i) { b[i] = lo[i]; b[8 + i] = hi[i]; }
    return b;
}

// ---------------- one-shot prep: transpose weights to bf16, densify rel bias ----------------
__global__ __launch_bounds__(256) void swin_prep_kernel(
    const float* __restrict__ qkv_w,      // [192][576]
    const float* __restrict__ proj_w,     // [192][192]
    const float* __restrict__ bias_table, // [169][6]
    const int*   __restrict__ rel_index,  // [49][49]
    __bf16* __restrict__ wTq,             // [576][192]  (row-major W^T)
    __bf16* __restrict__ wTp,             // [192][192]  (row-major Wp^T)
    float*  __restrict__ relb)            // [6][49][49]
{
    const int tid = blockIdx.x * 256 + threadIdx.x;
    const int stride = gridDim.x * 256;
    for (int i = tid; i < 576 * EDIM; i += stride) {
        int n = i / EDIM, k = i % EDIM;
        wTq[i] = (__bf16)qkv_w[k * 576 + n];
    }
    for (int i = tid; i < EDIM * EDIM; i += stride) {
        int n = i / EDIM, k = i % EDIM;
        wTp[i] = (__bf16)proj_w[k * EDIM + n];
    }
    for (int i = tid; i < N_HEADS * NTOK * NTOK; i += stride) {
        int h = i / (NTOK * NTOK), ij = i % (NTOK * NTOK);
        relb[i] = bias_table[rel_index[ij] * N_HEADS + h];
    }
}

// ---------------- fused window attention: one block (8 wave32) per window ----------------
__global__ __launch_bounds__(256) void swin_attn_kernel(
    const float* __restrict__ x,       // [B][49][192]
    const float* __restrict__ mask,    // [1024][49][49]
    const __bf16* __restrict__ wTq,    // [576][192]
    const float* __restrict__ qkv_b,   // [576]
    const __bf16* __restrict__ wTp,    // [192][192]
    const float* __restrict__ proj_b,  // [192]
    const float* __restrict__ relb,    // [6][49][49]
    float* __restrict__ out)           // [B][49][192]
{
    extern __shared__ char smem[];
    __bf16* XB = (__bf16*)(smem);            // [64][192] x (bf16), later attn output O
    __bf16* Qb = (__bf16*)(smem + 24576);    // [64][192] scaled q row-major
    __bf16* Kb = (__bf16*)(smem + 49152);    // [64][192] k row-major
    __bf16* VT = (__bf16*)(smem + 73728);    // [192][64] v transposed
    float*  Sb = (float*) (smem + 98304);    // [64][SPITCH] scores f32 (padded pitch)
    __bf16* Pb = (__bf16*)(smem + 115712);   // [64][64]  softmax probs bf16

    const int b    = blockIdx.x;
    const int tid  = threadIdx.x;
    const int wave = tid >> 5;
    const int lane = tid & 31;
    const int g    = lane >> 4;
    const int ln   = lane & 15;
    const float scale = 0.17677669529663688f;   // 1/sqrt(32)

    // ---- Phase 0: x -> bf16 LDS, 4 elems/iter (zero-pad rows 49..63) ----
    const float* xsrc = x + (size_t)b * (NTOK * EDIM);
    for (int i = tid * 4; i < NPAD * EDIM; i += 256 * 4) {
        int r = i / EDIM;
        v4bf pk;
        if (r < NTOK) {
            v4f v = *(const v4f*)(xsrc + i);
#pragma unroll
            for (int e = 0; e < 4; ++e) pk[e] = (__bf16)v[e];
        } else {
#pragma unroll
            for (int e = 0; e < 4; ++e) pk[e] = (__bf16)0.0f;
        }
        *(v4bf*)(XB + i) = pk;
    }
    __syncthreads();

    // ---- Phase 1: QKV^T = Wqkv^T @ X^T  (lane's 8 outputs = 8 consecutive channels) ----
    // wave owns token-strip mt = wave&3 (B frags hoisted); sweeps 18 channel-tiles.
    // Segment-specific loops (uniform ranges) -> no per-lane/per-iter segment compares.
    {
        const int mt = wave & 3;
        const int nh = wave >> 2;          // 0 or 1
        const int m0 = mt * 16;            // token tile base
        v16bf xfrag[6];
#pragma unroll
        for (int ks = 0; ks < 6; ++ks)
            xfrag[ks] = load_B_frag(XB, m0, ks * 32, EDIM, lane);

        const int tok = m0 + ln;           // this lane's token row

#define QKV_GEMM(N0, ACC)                                                          \
        do {                                                                       \
            _Pragma("unroll")                                                      \
            for (int ks = 0; ks < 6; ++ks) {                                       \
                v16bf wf = load_A_frag(wTq, (N0), ks * 32, EDIM, lane);            \
                (ACC) = __builtin_amdgcn_wmma_f32_16x16x32_bf16(                   \
                            false, wf, false, xfrag[ks], (short)0, (ACC), false, false); \
            }                                                                      \
        } while (0)

        if (nh == 0) {
            for (int nt = 0; nt < 12; ++nt) {        // ---- Q tiles (scaled) ----
                const int n0 = nt * 16;
                v8f acc = {};
                QKV_GEMM(n0, acc);
                const int nbase = n0 + g * 8;
                v4f b0 = *(const v4f*)(qkv_b + nbase);
                v4f b1 = *(const v4f*)(qkv_b + nbase + 4);
                v8bf pk;
#pragma unroll
                for (int r = 0; r < 8; ++r)
                    pk[r] = (__bf16)((acc[r] + (r < 4 ? b0[r] : b1[r - 4])) * scale);
                *(v8bf*)(Qb + tok * EDIM + nbase) = pk;
            }
            for (int nt = 12; nt < 18; ++nt) {       // ---- K tiles (first half) ----
                const int n0 = nt * 16;
                v8f acc = {};
                QKV_GEMM(n0, acc);
                const int nbase = n0 + g * 8;
                v4f b0 = *(const v4f*)(qkv_b + nbase);
                v4f b1 = *(const v4f*)(qkv_b + nbase + 4);
                v8bf pk;
#pragma unroll
                for (int r = 0; r < 8; ++r)
                    pk[r] = (__bf16)(acc[r] + (r < 4 ? b0[r] : b1[r - 4]));
                *(v8bf*)(Kb + tok * EDIM + (nbase - EDIM)) = pk;
            }
        } else {
            for (int nt = 18; nt < 24; ++nt) {       // ---- K tiles (second half) ----
                const int n0 = nt * 16;
                v8f acc = {};
                QKV_GEMM(n0, acc);
                const int nbase = n0 + g * 8;
                v4f b0 = *(const v4f*)(qkv_b + nbase);
                v4f b1 = *(const v4f*)(qkv_b + nbase + 4);
                v8bf pk;
#pragma unroll
                for (int r = 0; r < 8; ++r)
                    pk[r] = (__bf16)(acc[r] + (r < 4 ? b0[r] : b1[r - 4]));
                *(v8bf*)(Kb + tok * EDIM + (nbase - EDIM)) = pk;
            }
            for (int nt = 24; nt < 36; ++nt) {       // ---- V tiles -> transposed buffer ----
                const int n0 = nt * 16;
                v8f acc = {};
                QKV_GEMM(n0, acc);
                const int nbase = n0 + g * 8;
                v4f b0 = *(const v4f*)(qkv_b + nbase);
                v4f b1 = *(const v4f*)(qkv_b + nbase + 4);
                __bf16* vp = VT + (nbase - 2 * EDIM) * NPAD + tok;  // single base address
#pragma unroll
                for (int r = 0; r < 8; ++r)
                    vp[r * NPAD] = (__bf16)(acc[r] + (r < 4 ? b0[r] : b1[r - 4]));
            }
        }
#undef QKV_GEMM
    }
    __syncthreads();

    // ---- Phase 2/3: per-head attention ----
    const int mw = b & 1023;  // window b uses mask[b % 1024]
    for (int h = 0; h < N_HEADS; ++h) {
        // S = Qh @ Kh^T : 16 tiles, 2 per wave, single K=32 step
#pragma unroll
        for (int i = 0; i < 2; ++i) {
            const int t  = wave * 2 + i;
            const int m0 = (t >> 2) * 16;
            const int n0 = (t & 3) * 16;
            v16bf a  = load_A_frag(Qb, m0, h * HDIM, EDIM, lane);
            v16bf bb = load_B_frag(Kb, n0, h * HDIM, EDIM, lane);
            v8f acc = {};
            acc = __builtin_amdgcn_wmma_f32_16x16x32_bf16(false, a, false, bb,
                                                          (short)0, acc, false, false);
            float* sp = Sb + (m0 + g * 8) * SPITCH + n0 + ln;       // single base address
#pragma unroll
            for (int r = 0; r < 8; ++r)
                sp[r * SPITCH] = acc[r];
        }
        __syncthreads();

        // rowwise softmax: 4 lanes per row, register-resident, shfl_xor reductions
        {
            const int m   = tid >> 2;
            const int sub = tid & 3;
            float vreg[13];
            float mx = -1e30f;
            const float* rb = relb + ((h * NTOK) + m) * NTOK;
            const float* mk = mask + ((size_t)mw * NTOK + m) * NTOK;
            if (m < NTOK) {
#pragma unroll
                for (int t = 0; t < 13; ++t) {
                    const int j = sub + 4 * t;
                    float s = (j < NTOK) ? (Sb[m * SPITCH + j] + rb[j] + mk[j]) : -1e30f;
                    vreg[t] = s;
                    mx = fmaxf(mx, s);
                }
            }
            mx = fmaxf(mx, __shfl_xor(mx, 1));
            mx = fmaxf(mx, __shfl_xor(mx, 2));
            float sum = 0.0f;
            if (m < NTOK) {
#pragma unroll
                for (int t = 0; t < 13; ++t) {
                    const int j = sub + 4 * t;
                    float e = (j < NTOK) ? __expf(vreg[t] - mx) : 0.0f;
                    vreg[t] = e;
                    sum += e;
                }
            }
            sum += __shfl_xor(sum, 1);
            sum += __shfl_xor(sum, 2);
            if (m < NTOK) {
                const float inv = 1.0f / sum;
#pragma unroll
                for (int t = 0; t < 13; ++t) {
                    const int j = sub + 4 * t;
                    if (j < NTOK) Pb[m * NPAD + j] = (__bf16)(vreg[t] * inv);
                }
                for (int j = NTOK + sub; j < NPAD; j += 4)
                    Pb[m * NPAD + j] = (__bf16)0.0f;
            } else {
                for (int j = sub; j < NPAD; j += 4)
                    Pb[m * NPAD + j] = (__bf16)0.0f;
            }
        }
        __syncthreads();

        // O_h^T = V_h^T @ P^T : 8 tiles (2 d-tiles x 4 query-tiles), 1 per wave, 2 K steps.
        // Lane's 8 outputs = 8 consecutive channels of one query token -> packed b128 store.
        {
            const int d0  = (wave & 1) * 16;       // head-dim tile
            const int mq0 = (wave >> 1) * 16;      // query-token tile
            v8f acc = {};
#pragma unroll
            for (int ks = 0; ks < 2; ++ks) {
                v16bf a  = load_A_frag(VT + h * HDIM * NPAD, d0, ks * 32, NPAD, lane);
                v16bf bb = load_B_frag(Pb, mq0, ks * 32, NPAD, lane);
                acc = __builtin_amdgcn_wmma_f32_16x16x32_bf16(false, a, false, bb,
                                                              (short)0, acc, false, false);
            }
            v8bf pk;
#pragma unroll
            for (int r = 0; r < 8; ++r) pk[r] = (__bf16)acc[r];
            *(v8bf*)(XB + (mq0 + ln) * EDIM + h * HDIM + d0 + g * 8) = pk;
        }
        // barrier at top of next iteration (before softmax) protects Pb/Sb reuse
    }
    __syncthreads();

    // ---- Phase 4: out^T = Wp^T @ O^T + b : packed global b128 stores ----
    {
        const int mt = wave & 3;
        const int nh = wave >> 2;
        const int m0 = mt * 16;                // token tile base
        v16bf ofrag[6];
#pragma unroll
        for (int ks = 0; ks < 6; ++ks)
            ofrag[ks] = load_B_frag(XB, m0, ks * 32, EDIM, lane);

        float* osrc = out + (size_t)b * (NTOK * EDIM);
        const int tok = m0 + ln;
        for (int nt = nh * 6; nt < nh * 6 + 6; ++nt) {
            const int n0 = nt * 16;
            v8f acc = {};
#pragma unroll
            for (int ks = 0; ks < 6; ++ks) {
                v16bf wf = load_A_frag(wTp, n0, ks * 32, EDIM, lane);
                acc = __builtin_amdgcn_wmma_f32_16x16x32_bf16(false, wf, false, ofrag[ks],
                                                              (short)0, acc, false, false);
            }
            const int nbase = n0 + g * 8;
            v4f b0 = *(const v4f*)(proj_b + nbase);
            v4f b1 = *(const v4f*)(proj_b + nbase + 4);
            if (tok < NTOK) {
                v4f o0, o1;
#pragma unroll
                for (int r = 0; r < 4; ++r) { o0[r] = acc[r] + b0[r]; o1[r] = acc[4 + r] + b1[r]; }
                *(v4f*)(osrc + tok * EDIM + nbase)     = o0;
                *(v4f*)(osrc + tok * EDIM + nbase + 4) = o1;
            }
        }
    }
}

extern "C" void kernel_launch(void* const* d_in, const int* in_sizes, int n_in,
                              void* d_out, int out_size, void* d_ws, size_t ws_size,
                              hipStream_t stream) {
    (void)in_sizes; (void)n_in; (void)out_size; (void)ws_size;
    const float* x          = (const float*)d_in[0];
    const float* mask       = (const float*)d_in[1];
    const float* qkv_w      = (const float*)d_in[2];
    const float* qkv_b      = (const float*)d_in[3];
    const float* proj_w     = (const float*)d_in[4];
    const float* proj_b     = (const float*)d_in[5];
    const float* bias_table = (const float*)d_in[6];
    const int*   rel_index  = (const int*)d_in[7];

    char* ws = (char*)d_ws;
    __bf16* wTq  = (__bf16*)(ws);                                   // 576*192*2 = 221184 B
    __bf16* wTp  = (__bf16*)(ws + 576 * 192 * 2);                   // 192*192*2 =  73728 B
    float*  relb = (float*) (ws + 576 * 192 * 2 + 192 * 192 * 2);   // 6*49*49*4 =  57624 B

    swin_prep_kernel<<<432, 256, 0, stream>>>(qkv_w, proj_w, bias_table, rel_index,
                                              wTq, wTp, relb);

    const int SMEM_BYTES = 123904;  // 121 KB of the 320 KB WGP LDS
    hipFuncSetAttribute(reinterpret_cast<const void*>(swin_attn_kernel),
                        hipFuncAttributeMaxDynamicSharedMemorySize, SMEM_BYTES);
    swin_attn_kernel<<<8192, 256, SMEM_BYTES, stream>>>(
        x, mask, wTq, qkv_b, wTp, proj_b, relb, (float*)d_out);
}
